// InteractionPredictor_45509473468604
// MI455X (gfx1250) — matrix operand order repaired and checked by reading, not compiled, not run on hardware
//
#include <hip/hip_runtime.h>
#include <hip/hip_bf16.h>
#include <math.h>

#define N_NODES  524288
#define N_EDGES  2097152
#define N_GRAPHS 16384
#define COMP_DIM 3539
#define PROT_DIM 384
#define GNN_HID  64
#define G_EMB    128
#define HID      256
#define WPB      4   // waves per block in GEMM
#define MT       4   // M-tiles per wave (B fragment reused MT times)

typedef __bf16 bf16_t;
typedef bf16_t v16bf __attribute__((ext_vector_type(16)));
typedef float  v8f   __attribute__((ext_vector_type(8)));

// ---------------------------------------------------------------------------
// Utility kernels
// ---------------------------------------------------------------------------
__global__ void zero_f32(float* __restrict__ p, long long n) {
    long long i = (long long)blockIdx.x * blockDim.x + threadIdx.x;
    if (i < n) p[i] = 0.0f;
}

__global__ void deg_init(float* __restrict__ deg) {
    int i = blockIdx.x * blockDim.x + threadIdx.x;
    if (i < N_NODES) deg[i] = 1.0f;   // self-loop contributes 1
}

__global__ void deg_accum(const int* __restrict__ col, float* __restrict__ deg) {
    int e = blockIdx.x * blockDim.x + threadIdx.x;
    if (e < N_EDGES) atomicAdd(&deg[col[e]], 1.0f);
}

__global__ void deg_rsqrt(float* __restrict__ deg) {
    int i = blockIdx.x * blockDim.x + threadIdx.x;
    if (i < N_NODES) {
        float d = deg[i];
        deg[i] = (d > 0.0f) ? rsqrtf(d) : 0.0f;   // becomes dinv
    }
}

// ---------------------------------------------------------------------------
// GCN layer 1: h = x(N,1) @ W1(1,64); acc[col] += norm * h[row]
// ---------------------------------------------------------------------------
__global__ void l1_scatter(const int* __restrict__ row, const int* __restrict__ col,
                           const float* __restrict__ dinv, const float* __restrict__ x,
                           const float* __restrict__ W1, float* __restrict__ acc) {
    long long t = (long long)blockIdx.x * blockDim.x + threadIdx.x;
    int e = (int)(t >> 6);
    int j = (int)(t & 63);
    if (e >= N_EDGES) return;
    int r = row[e], c = col[e];
    float w = dinv[r] * dinv[c] * x[r];
    atomicAdd(&acc[(size_t)c * GNN_HID + j], w * W1[j]);
}

// self-loop + bias + relu, in place
__global__ void l1_finish(const float* __restrict__ x, const float* __restrict__ dinv,
                          const float* __restrict__ W1, const float* __restrict__ b1,
                          float* __restrict__ acc) {
    long long t = (long long)blockIdx.x * blockDim.x + threadIdx.x;
    if (t >= (long long)N_NODES * GNN_HID) return;
    int n = (int)(t >> 6), j = (int)(t & 63);
    float d = dinv[n];
    float v = acc[t] + d * d * x[n] * W1[j] + b1[j];
    acc[t] = fmaxf(v, 0.0f);
}

// ---------------------------------------------------------------------------
// GCN layer 2 scatter: acc[col] += norm * t2[row]
// ---------------------------------------------------------------------------
__global__ void l2_scatter(const int* __restrict__ row, const int* __restrict__ col,
                           const float* __restrict__ dinv, const float* __restrict__ t2,
                           float* __restrict__ acc) {
    long long t = (long long)blockIdx.x * blockDim.x + threadIdx.x;
    int e = (int)(t >> 6);
    int j = (int)(t & 63);
    if (e >= N_EDGES) return;
    int r = row[e], c = col[e];
    float w = dinv[r] * dinv[c];
    atomicAdd(&acc[(size_t)c * GNN_HID + j], w * t2[(size_t)r * GNN_HID + j]);
}

__global__ void l2_finish(const float* __restrict__ dinv, const float* __restrict__ t2,
                          const float* __restrict__ b2, float* __restrict__ acc) {
    long long t = (long long)blockIdx.x * blockDim.x + threadIdx.x;
    if (t >= (long long)N_NODES * GNN_HID) return;
    int n = (int)(t >> 6), j = (int)(t & 63);
    float d = dinv[n];
    float v = acc[t] + d * d * t2[t] + b2[j];
    acc[t] = fmaxf(v, 0.0f);
}

// ---------------------------------------------------------------------------
// Global mean pool
// ---------------------------------------------------------------------------
__global__ void pool_accum(const float* __restrict__ x2, const int* __restrict__ batch,
                           float* __restrict__ sums, float* __restrict__ cnt) {
    long long t = (long long)blockIdx.x * blockDim.x + threadIdx.x;
    if (t >= (long long)N_NODES * GNN_HID) return;
    int n = (int)(t >> 6), j = (int)(t & 63);
    int b = batch[n];
    atomicAdd(&sums[(size_t)b * GNN_HID + j], x2[t]);
    if (j == 0) atomicAdd(&cnt[b], 1.0f);
}

__global__ void pool_div(const float* __restrict__ sums, const float* __restrict__ cnt,
                         float* __restrict__ pooled) {
    int t = blockIdx.x * blockDim.x + threadIdx.x;
    if (t >= N_GRAPHS * GNN_HID) return;
    int b = t >> 6;
    pooled[t] = sums[t] / fmaxf(cnt[b], 1.0f);
}

// ---------------------------------------------------------------------------
// Fragment packing: convert f32 row-major A/B into fragment-major bf16 so the
// WMMA GEMM inner loop is pure 128-bit loads + v_wmma (no bounds checks, no
// cvt, no divergence). Fragment f = tile*KC + kc, laid out [32 lanes][16 bf16].
//   A (16x32, M-tile): lanes 0-15 row l, K {0..7,16..23}; lanes 16-31 K {8..15,24..31}
//   B (32x16, N-tile): lane = column; lanes 0-15 K {0..7,16..23}; lanes 16-31 K {8..15,24..31}
// K tails are zero-filled here, so the GEMM needs no K guard.
// ---------------------------------------------------------------------------
__global__ void pack_a(const float* __restrict__ A, bf16_t* __restrict__ pA,
                       int M, int K, int lda, int KC, long long total) {
    long long t = (long long)blockIdx.x * blockDim.x + threadIdx.x;
    if (t >= total) return;
    int i    = (int)(t & 15);
    int lane = (int)((t >> 4) & 31);
    long long f = t >> 9;
    int kc    = (int)(f % KC);
    int mtile = (int)(f / KC);
    int half = lane >> 4, l = lane & 15;
    int m  = mtile * 16 + l;
    int kk = kc * 32 + half * 8 + (i < 8 ? i : i + 8);
    float v = (m < M && kk < K) ? A[(size_t)m * lda + kk] : 0.0f;
    pA[t] = (bf16_t)v;
}

__global__ void pack_b(const float* __restrict__ B, bf16_t* __restrict__ pB,
                       int K, int N, int ldb, int KC, long long total) {
    long long t = (long long)blockIdx.x * blockDim.x + threadIdx.x;
    if (t >= total) return;
    int i    = (int)(t & 15);
    int lane = (int)((t >> 4) & 31);
    long long f = t >> 9;
    int kc    = (int)(f % KC);
    int ntile = (int)(f / KC);
    int half = lane >> 4, l = lane & 15;
    int n  = ntile * 16 + l;
    int kk = kc * 32 + half * 8 + (i < 8 ? i : i + 8);
    float v = (kk < K && n < N) ? B[(size_t)kk * ldb + n] : 0.0f;
    pB[t] = (bf16_t)v;
}

// ---------------------------------------------------------------------------
// WMMA GEMM on packed fragments: C(M,N) {=, +=} A @ B [+bias] [relu]
// One wave computes MT consecutive 16x16 M-tiles for one N-tile, so each B
// fragment load is reused MT times (4x less B traffic, 4 WMMAs per K-chunk).
// All launches here have M % (16*MT) == 0 and N % 16 == 0: no edge guards.
// ---------------------------------------------------------------------------
__global__ void gemm_wmma_packed(const v16bf* __restrict__ pA, const v16bf* __restrict__ pB,
                                 float* __restrict__ C, const float* __restrict__ bias,
                                 int Mtiles, int KC, int ldc,
                                 int accumulate, int relu) {
    const int lane = threadIdx.x & 31;
    const int wave = threadIdx.x >> 5;
    const int tn  = blockIdx.x;
    const int tm0 = (blockIdx.y * (blockDim.x >> 5) + wave) * MT;
    if (tm0 >= Mtiles) return;   // wave-uniform: EXEC all-ones at WMMA

    const v16bf* ap = pA + ((size_t)tm0 * KC) * 32 + lane;   // MT tiles, KC*32 apart
    const v16bf* bp = pB + ((size_t)tn  * KC) * 32 + lane;

    const int half = lane >> 4;
    const int l    = lane & 15;
    const int n    = tn * 16 + l;
    float* Cp = C + (size_t)(tm0 * 16 + half * 8) * ldc + n;

    v8f acc[MT];
#pragma unroll
    for (int t = 0; t < MT; ++t) {
        v8f z = {0.f, 0.f, 0.f, 0.f, 0.f, 0.f, 0.f, 0.f};
        acc[t] = z;
    }
    if (accumulate) {
#pragma unroll
        for (int t = 0; t < MT; ++t)
#pragma unroll
            for (int r = 0; r < 8; ++r)
                acc[t][r] = Cp[(size_t)(t * 16 + r) * ldc];
    }

#pragma unroll 2
    for (int kc = 0; kc < KC; ++kc) {
        __builtin_prefetch(ap + (size_t)(kc + 4) * 32, 0, 0);   // speculative, safe OOB
        v16bf b = bp[(size_t)kc * 32];
        v16bf a0 = ap[(size_t)kc * 32];
        v16bf a1 = ap[(size_t)(KC + kc) * 32];
        v16bf a2 = ap[(size_t)(2 * KC + kc) * 32];
        v16bf a3 = ap[(size_t)(3 * KC + kc) * 32];
        acc[0] = __builtin_amdgcn_wmma_f32_16x16x32_bf16(false, a0, false, b, (short)0, acc[0], false, false);
        acc[1] = __builtin_amdgcn_wmma_f32_16x16x32_bf16(false, a1, false, b, (short)0, acc[1], false, false);
        acc[2] = __builtin_amdgcn_wmma_f32_16x16x32_bf16(false, a2, false, b, (short)0, acc[2], false, false);
        acc[3] = __builtin_amdgcn_wmma_f32_16x16x32_bf16(false, a3, false, b, (short)0, acc[3], false, false);
    }

    float bv = (bias != nullptr) ? bias[n] : 0.0f;
#pragma unroll
    for (int t = 0; t < MT; ++t) {
#pragma unroll
        for (int r = 0; r < 8; ++r) {
            float v = acc[t][r] + bv;
            if (relu) v = fmaxf(v, 0.0f);
            Cp[(size_t)(t * 16 + r) * ldc] = v;
        }
    }
}

// ---------------------------------------------------------------------------
// Head: out = sigmoid(h @ Wf2 + bf2)   (N=1 -> wave-shuffle reduction)
// ---------------------------------------------------------------------------
__global__ void final_head(const float* __restrict__ h, const float* __restrict__ Wf2,
                           const float* __restrict__ bf2, float* __restrict__ out) {
    int wave = threadIdx.x >> 5;
    int lane = threadIdx.x & 31;
    int r = blockIdx.x * (blockDim.x >> 5) + wave;
    if (r >= N_GRAPHS) return;
    const float* hr = h + (size_t)r * HID;
    float s = 0.0f;
#pragma unroll
    for (int j = lane; j < HID; j += 32) s += hr[j] * Wf2[j];
#pragma unroll
    for (int off = 16; off > 0; off >>= 1) s += __shfl_xor(s, off, 32);
    if (lane == 0) out[r] = 1.0f / (1.0f + __expf(-(s + bf2[0])));
}

// ---------------------------------------------------------------------------
static inline unsigned gridFor(long long n, int b) { return (unsigned)((n + b - 1) / b); }

// pack A, pack B, then run the WMMA GEMM
static void launch_gemm(const float* A, const float* B, float* C, const float* bias,
                        int M, int N, int K, int lda, int ldb, int ldc,
                        int accumulate, int relu,
                        bf16_t* packA, bf16_t* packB, hipStream_t stream) {
    int Mtiles = (M + 15) / 16;
    int Ntiles = (N + 15) / 16;
    int KC     = (K + 31) / 32;
    long long na = (long long)Mtiles * KC * 512;
    long long nb = (long long)Ntiles * KC * 512;
    pack_a<<<gridFor(na, 256), dim3(256), 0, stream>>>(A, packA, M, K, lda, KC, na);
    pack_b<<<gridFor(nb, 256), dim3(256), 0, stream>>>(B, packB, K, N, ldb, KC, nb);
    dim3 grid(Ntiles, gridFor(Mtiles, WPB * MT));
    gemm_wmma_packed<<<grid, dim3(32 * WPB), 0, stream>>>(
        (const v16bf*)packA, (const v16bf*)packB, C, bias, Mtiles, KC, ldc,
        accumulate, relu);
}

extern "C" void kernel_launch(void* const* d_in, const int* in_sizes, int n_in,
                              void* d_out, int out_size, void* d_ws, size_t ws_size,
                              hipStream_t stream) {
    const float* node_x = (const float*)d_in[0];
    const float* comp   = (const float*)d_in[1];
    const float* prot   = (const float*)d_in[2];
    const int*   eidx   = (const int*)d_in[3];
    const int*   batch  = (const int*)d_in[4];
    const float* W1  = (const float*)d_in[5];
    const float* b1  = (const float*)d_in[6];
    const float* W2  = (const float*)d_in[7];
    const float* b2  = (const float*)d_in[8];
    const float* Wg  = (const float*)d_in[9];
    const float* bg  = (const float*)d_in[10];
    const float* Wf1 = (const float*)d_in[11];
    const float* bf1 = (const float*)d_in[12];
    const float* Wf2 = (const float*)d_in[13];
    const float* bf2 = (const float*)d_in[14];
    float* out = (float*)d_out;

    // Workspace layout
    float* ws     = (float*)d_ws;
    float* dinv   = ws;                                     // N_NODES
    float* accA   = dinv + N_NODES;                         // N_NODES*64 (scatter acc / x1 / x2)
    float* accB   = accA + (size_t)N_NODES * GNN_HID;       // N_NODES*64 (t2 = x1@W2)
    float* sums   = accB + (size_t)N_NODES * GNN_HID;       // N_GRAPHS*64
    float* cnt    = sums + (size_t)N_GRAPHS * GNN_HID;      // N_GRAPHS
    float* pooled = cnt  + N_GRAPHS;                        // N_GRAPHS*64
    float* gbuf   = pooled + (size_t)N_GRAPHS * GNN_HID;    // N_GRAPHS*128
    float* hbuf   = gbuf + (size_t)N_GRAPHS * G_EMB;        // N_GRAPHS*256
    // bf16 pack regions (time-shared across GEMMs):
    //   packA worst case: comp GEMM -> 1024 tiles * 111 KC * 512 = 58.2M bf16
    //   packB worst case: comp GEMM -> 16 tiles * 111 KC * 512 ~= 0.91M bf16
    bf16_t* packA = (bf16_t*)(hbuf + (size_t)N_GRAPHS * HID);
    bf16_t* packB = packA + ((size_t)64 * 1024 * 1024);     // 64M bf16 headroom

    const int* erow = eidx;
    const int* ecol = eidx + N_EDGES;

    const dim3 B256(256);
    const long long accN = (long long)N_NODES * GNN_HID;
    const long long edgT = (long long)N_EDGES * GNN_HID;

    // ---- GCN normalization -------------------------------------------------
    deg_init <<<gridFor(N_NODES, 256), B256, 0, stream>>>(dinv);
    deg_accum<<<gridFor(N_EDGES, 256), B256, 0, stream>>>(ecol, dinv);
    deg_rsqrt<<<gridFor(N_NODES, 256), B256, 0, stream>>>(dinv);

    // ---- GCN layer 1 -------------------------------------------------------
    zero_f32  <<<gridFor(accN, 256), B256, 0, stream>>>(accA, accN);
    l1_scatter<<<gridFor(edgT, 256), B256, 0, stream>>>(erow, ecol, dinv, node_x, W1, accA);
    l1_finish <<<gridFor(accN, 256), B256, 0, stream>>>(node_x, dinv, W1, b1, accA);

    // ---- GCN layer 2: t2 = x1 @ W2 (WMMA), then scatter --------------------
    launch_gemm(accA, W2, accB, nullptr, N_NODES, GNN_HID, GNN_HID,
                GNN_HID, GNN_HID, GNN_HID, 0, 0, packA, packB, stream);
    zero_f32  <<<gridFor(accN, 256), B256, 0, stream>>>(accA, accN);
    l2_scatter<<<gridFor(edgT, 256), B256, 0, stream>>>(erow, ecol, dinv, accB, accA);
    l2_finish <<<gridFor(accN, 256), B256, 0, stream>>>(dinv, accB, b2, accA);

    // ---- Global mean pool --------------------------------------------------
    zero_f32  <<<gridFor((long long)N_GRAPHS * GNN_HID + N_GRAPHS, 256), B256, 0, stream>>>(
        sums, (long long)N_GRAPHS * GNN_HID + N_GRAPHS);   // sums + cnt contiguous
    pool_accum<<<gridFor(accN, 256), B256, 0, stream>>>(accA, batch, sums, cnt);
    pool_div  <<<gridFor((long long)N_GRAPHS * GNN_HID, 256), B256, 0, stream>>>(sums, cnt, pooled);

    // ---- g = pooled @ Wg + bg ---------------------------------------------
    launch_gemm(pooled, Wg, gbuf, bg, N_GRAPHS, G_EMB, GNN_HID,
                GNN_HID, G_EMB, G_EMB, 0, 0, packA, packB, stream);

    // ---- h = relu([g|comp|prot] @ Wf1 + bf1), K-split into 3 GEMMs --------
    // rows 0..127 of Wf1 <- g
    launch_gemm(gbuf, Wf1, hbuf, bf1, N_GRAPHS, HID, G_EMB,
                G_EMB, HID, HID, 0, 0, packA, packB, stream);
    // rows 128..3666 <- compound_feat
    launch_gemm(comp, Wf1 + (size_t)G_EMB * HID, hbuf, nullptr, N_GRAPHS, HID, COMP_DIM,
                COMP_DIM, HID, HID, 1, 0, packA, packB, stream);
    // rows 3667..4050 <- protein_feat, final accumulate + relu
    launch_gemm(prot, Wf1 + (size_t)(G_EMB + COMP_DIM) * HID, hbuf, nullptr, N_GRAPHS, HID, PROT_DIM,
                PROT_DIM, HID, HID, 1, 1, packA, packB, stream);

    // ---- out = sigmoid(h @ Wf2 + bf2) -------------------------------------
    final_head<<<gridFor(N_GRAPHS, 256 / 32), dim3(256), 0, stream>>>(hbuf, Wf2, bf2, out);
}
